// FP8DynamicMatmul_10574209483056
// MI455X (gfx1250) — 8-bit matrix operand order repaired and checked by reading, compiled, not run on hardware
//
#include <hip/hip_runtime.h>
#include <stdint.h>

#define MDIM 8192
#define KDIM 4096
#define NDIM 4096
#define FP8_MAX 448.0f
#define KTILES (KDIM / 128)

typedef __attribute__((ext_vector_type(16))) int   v16i;
typedef __attribute__((ext_vector_type(8)))  float v8f;


__device__ __forceinline__ float wave_max32(float m) {
#pragma unroll
  for (int off = 16; off > 0; off >>= 1)
    m = fmaxf(m, __shfl_xor(m, off, 32));
  return m;
}

__global__ void init_ws_kernel(unsigned* amax) {
  if (threadIdx.x < 2) amax[threadIdx.x] = 0u;
}

__global__ void amax_kernel(const float4* __restrict__ x, size_t n4,
                            unsigned* __restrict__ out) {
  float m = 0.0f;
  for (size_t i = (size_t)blockIdx.x * blockDim.x + threadIdx.x; i < n4;
       i += (size_t)gridDim.x * blockDim.x) {
    float4 v = x[i];
    m = fmaxf(m, fmaxf(fmaxf(fabsf(v.x), fabsf(v.y)),
                       fmaxf(fabsf(v.z), fabsf(v.w))));
  }
  m = wave_max32(m);
  if ((threadIdx.x & 31) == 0) atomicMax(out, __float_as_uint(m));
}

__device__ __forceinline__ unsigned char f32_to_e4m3_sw(float x) {
  unsigned bits = __float_as_uint(x);
  unsigned sign = (bits >> 24) & 0x80u;
  float a = fabsf(x);
  if (a >= FP8_MAX) return (unsigned char)(sign | 0x7Eu);
  if (a < 0.0009765625f) {
    unsigned q = (unsigned)rintf(ldexpf(a, 9));
    if (q > 7u) return (unsigned char)(sign | 0x08u);
    return (unsigned char)(sign | q);
  }
  int e;
  (void)frexpf(a, &e);
  float step = ldexpf(1.0f, e - 4);
  unsigned q = (unsigned)rintf(a / step);
  if (q == 16u) { q = 8u; e++; }
  int expf8 = e - 1 + 7;
  if (expf8 >= 16) return (unsigned char)(sign | 0x7Eu);
  if (expf8 <= 0) {
    unsigned qq = (unsigned)rintf(ldexpf(a, 9));
    if (qq > 7u) return (unsigned char)(sign | 0x08u);
    return (unsigned char)(sign | qq);
  }
  return (unsigned char)(sign | ((unsigned)expf8 << 3) | (q & 7u));
}

__device__ __forceinline__ unsigned pack4_fp8(float x0, float x1, float x2,
                                              float x3) {
  x0 = fminf(fmaxf(x0, -FP8_MAX), FP8_MAX);
  x1 = fminf(fmaxf(x1, -FP8_MAX), FP8_MAX);
  x2 = fminf(fmaxf(x2, -FP8_MAX), FP8_MAX);
  x3 = fminf(fmaxf(x3, -FP8_MAX), FP8_MAX);
#if defined(__gfx1250__) && __has_builtin(__builtin_amdgcn_cvt_pk_fp8_f32)
  int v = 0;
  v = __builtin_amdgcn_cvt_pk_fp8_f32(x0, x1, v, false);
  v = __builtin_amdgcn_cvt_pk_fp8_f32(x2, x3, v, true);
  return (unsigned)v;
#else
  return (unsigned)f32_to_e4m3_sw(x0) | ((unsigned)f32_to_e4m3_sw(x1) << 8) |
         ((unsigned)f32_to_e4m3_sw(x2) << 16) |
         ((unsigned)f32_to_e4m3_sw(x3) << 24);
#endif
}

__device__ __forceinline__ size_t a_swizzle_off(int m, int k) {
  int tile = (m >> 4) * KTILES + (k >> 7);
  int lane = (m & 15) | (((k >> 3) & 1) << 4);
  int p = ((k & 63) >> 4) | (((k & 127) >> 6) << 2);
  return ((size_t)tile << 11) + (lane << 6) + (p << 3) + (k & 7);
}

__device__ __forceinline__ size_t b_swizzle_off(int n, int k) {
  int tile = (n >> 4) * KTILES + (k >> 7);
  int lane = (n & 15) | (((k >> 3) & 1) << 4);
  int c = (k & 127) >> 4;
  return ((size_t)tile << 11) + (lane << 6) + (c << 3) + (k & 7);
}

__global__ void quantA_kernel(const float4* __restrict__ A,
                              uint8_t* __restrict__ Aq,
                              const unsigned* __restrict__ amax, size_t n4) {
  size_t i = (size_t)blockIdx.x * blockDim.x + threadIdx.x;
  if (i >= n4) return;
  float s_inv = FP8_MAX / fmaxf(__uint_as_float(amax[0]), 1e-30f);
  float4 v = A[i];
  int m = (int)(i >> 10);
  int k = (int)(i & 1023) << 2;
  unsigned w = pack4_fp8(v.x * s_inv, v.y * s_inv, v.z * s_inv, v.w * s_inv);
  *reinterpret_cast<unsigned*>(Aq + a_swizzle_off(m, k)) = w;
}

__global__ void quantB_kernel(const float* __restrict__ B,
                              uint8_t* __restrict__ Bq,
                              const unsigned* __restrict__ amax) {
  int n = blockIdx.x * 128 + threadIdx.x;
  int k0 = blockIdx.y * 4;
  float s_inv = FP8_MAX / fmaxf(__uint_as_float(amax[0]), 1e-30f);
  float x0 = B[(size_t)(k0 + 0) * NDIM + n] * s_inv;
  float x1 = B[(size_t)(k0 + 1) * NDIM + n] * s_inv;
  float x2 = B[(size_t)(k0 + 2) * NDIM + n] * s_inv;
  float x3 = B[(size_t)(k0 + 3) * NDIM + n] * s_inv;
  *reinterpret_cast<unsigned*>(Bq + b_swizzle_off(n, k0)) =
      pack4_fp8(x0, x1, x2, x3);
}

__global__ __launch_bounds__(256) void gemm_fp8_kernel(
    const uint8_t* __restrict__ Aq, const uint8_t* __restrict__ Bq,
    float* __restrict__ C, const unsigned* __restrict__ amax) {
  const int lane = threadIdx.x & 31;
  const int wave = threadIdx.x >> 5;
  const int wm = wave & 3;
  const int wn = wave >> 2;
  const int m0 = blockIdx.y * 256 + wm * 64;
  const int n0 = blockIdx.x * 128 + wn * 64;
  const int l15 = lane & 15;
  const int half = lane >> 4;

  const uint8_t* ap[4];
  const uint8_t* bp[4];
#pragma unroll
  for (int i = 0; i < 4; i++) {
    ap[i] = Aq + (((size_t)((m0 >> 4) + i) * KTILES) << 11) + (lane << 6);
    bp[i] = Bq + (((size_t)((n0 >> 4) + i) * KTILES) << 11) + (lane << 6);
  }

  v8f acc[4][4];
#pragma unroll
  for (int i = 0; i < 4; i++)
#pragma unroll
    for (int j = 0; j < 4; j++) acc[i][j] = (v8f){};

  for (int kc = 0; kc < KTILES; kc++) {
    const size_t koff = (size_t)kc << 11;
    v16i a[4];
#pragma unroll
    for (int i = 0; i < 4; i++) {
#pragma unroll
      for (int c = 0; c < 4; c++) {
        int4 w = *reinterpret_cast<const int4*>(ap[i] + koff + c * 16);
        a[i][4 * c + 0] = w.x; a[i][4 * c + 1] = w.y;
        a[i][4 * c + 2] = w.z; a[i][4 * c + 3] = w.w;
      }
    }
#pragma unroll
    for (int j = 0; j < 4; j++) {
      v16i b;
#pragma unroll
      for (int c = 0; c < 4; c++) {
        int4 w = *reinterpret_cast<const int4*>(bp[j] + koff + c * 16);
        b[4 * c + 0] = w.x; b[4 * c + 1] = w.y;
        b[4 * c + 2] = w.z; b[4 * c + 3] = w.w;
      }
#pragma unroll
      for (int i = 0; i < 4; i++) {
        acc[i][j] = __builtin_amdgcn_wmma_f32_16x16x128_fp8_fp8(
            a[i], b, (short)0, acc[i][j], false, false);
      }
    }
  }

  const float sA = __uint_as_float(amax[0]) * (1.0f / FP8_MAX);
  const float sB = __uint_as_float(amax[1]) * (1.0f / FP8_MAX);
  const float s = sA * sB;

  const int col0 = n0 + l15;
#pragma unroll
  for (int i = 0; i < 4; i++) {
    const int rowb = m0 + i * 16 + half * 8;
#pragma unroll
    for (int r = 0; r < 8; r++) {
      float* crow = C + (size_t)(rowb + r) * NDIM + col0;
#pragma unroll
      for (int j = 0; j < 4; j++) crow[j * 16] = acc[i][j][r] * s;
    }
  }
}

extern "C" void kernel_launch(void* const* d_in, const int* in_sizes, int n_in,
                              void* d_out, int out_size, void* d_ws,
                              size_t ws_size, hipStream_t stream) {
  (void)in_sizes; (void)n_in; (void)out_size; (void)ws_size;
  const float* A = (const float*)d_in[0];
  const float* B = (const float*)d_in[1];
  float* C = (float*)d_out;

  unsigned* amax = (unsigned*)d_ws;
  uint8_t* Aq = (uint8_t*)d_ws + 256;
  uint8_t* Bq = Aq + (size_t)MDIM * KDIM;

  init_ws_kernel<<<1, 32, 0, stream>>>(amax);

  amax_kernel<<<2048, 256, 0, stream>>>((const float4*)A,
                                        (size_t)MDIM * KDIM / 4, amax + 0);
  amax_kernel<<<2048, 256, 0, stream>>>((const float4*)B,
                                        (size_t)KDIM * NDIM / 4, amax + 1);

  quantA_kernel<<<(MDIM * (size_t)KDIM / 4 + 255) / 256, 256, 0, stream>>>(
      (const float4*)A, Aq, amax + 0, (size_t)MDIM * KDIM / 4);
  quantB_kernel<<<dim3(NDIM / 128, KDIM / 4), dim3(128, 1), 0, stream>>>(
      B, Bq, amax + 1);

  gemm_fp8_kernel<<<dim3(NDIM / 128, MDIM / 256), 256, 0, stream>>>(Aq, Bq, C,
                                                                    amax);
}
